// MultiModalFusionBlock_71906342470084
// MI455X (gfx1250) — matrix operand order repaired and checked by reading, not compiled, not run on hardware
//
#include <hip/hip_runtime.h>
#include <hip/hip_bf16.h>

typedef __attribute__((ext_vector_type(16))) _Float16 v16h;
typedef __attribute__((ext_vector_type(8)))  _Float16 v8h;
typedef __attribute__((ext_vector_type(8)))  float    v8f;

union AFrag { v16h v; v8h h[2]; };

// ---------------------------------------------------------------------------
// Generic batched WMMA GEMM:  C[z] = A[z] (MxK, f16 row-major) * Bt[z]^T + bias
// Bt is stored [N x K] (B transposed) so both operand fragments are contiguous
// along K.  One wave computes a 32x64 tile: 2 A fragments x 4 B fragments ->
// 8 v_wmma_f32_16x16x32_f16 per 32-wide K step (12 b128 loads : 8 WMMAs).
// A 256-thread block (8 waves, 4 along M x 2 along N) covers a 128x128 tile.
// Fragment layouts used here (wave32); A and B both use contiguous K = 16*(lane/16) + s:
//   A (16x32 f16): lane%16 = row M; lane/16 selects K half {0..15} vs
//                  {16..31}  -> two 16-byte loads per lane.
//   B (32x16 f16): lane%16 = col N; lane/16 selects K half {0..15}/{16..31}
//                  -> one contiguous 32-byte load per lane.
//   C/D (16x16 f32): element r -> M = r + 8*(lane/16), N = lane%16.
// ---------------------------------------------------------------------------
__global__ __launch_bounds__(256) void wmma_gemm_kernel(
    const _Float16* __restrict__ A, const _Float16* __restrict__ Bt,
    float* __restrict__ C, const float* __restrict__ bias,
    int lda, int ldb, int ldc, int M, int N, int K,
    long sAb, long sAh, long sBb, long sBh, long sCb, long sCh, int Hn)
{
    const int zb = blockIdx.z / Hn;
    const int zh = blockIdx.z % Hn;
    A  += (long)zb * sAb + (long)zh * sAh;
    Bt += (long)zb * sBb + (long)zh * sBh;
    C  += (long)zb * sCb + (long)zh * sCh;

    const int lane = threadIdx.x & 31;
    const int wave = threadIdx.x >> 5;
    const int m0 = blockIdx.y * 128 + (wave >> 1) * 32;
    const int n0 = blockIdx.x * 128 + (wave & 1) * 64;
    if (m0 >= M || n0 >= N) return;

    const int r  = lane & 15;
    const int hi = lane >> 4;

    const _Float16* arow0 = A + (long)(m0 + r) * lda;
    const _Float16* arow1 = A + (long)(m0 + 16 + r) * lda;
    const _Float16* brow0 = Bt + (long)(n0 + r) * ldb;
    const _Float16* brow1 = brow0 + (long)16 * ldb;
    const _Float16* brow2 = brow0 + (long)32 * ldb;
    const _Float16* brow3 = brow0 + (long)48 * ldb;

    v8f acc00 = {}, acc01 = {}, acc02 = {}, acc03 = {};
    v8f acc10 = {}, acc11 = {}, acc12 = {}, acc13 = {};

    for (int k0 = 0; k0 < K; k0 += 32) {
        AFrag a0, a1;
        a0.h[0] = *(const v8h*)(arow0 + k0 + 16 * hi);
        a0.h[1] = *(const v8h*)(arow0 + k0 + 16 * hi + 8);
        a1.h[0] = *(const v8h*)(arow1 + k0 + 16 * hi);
        a1.h[1] = *(const v8h*)(arow1 + k0 + 16 * hi + 8);
        const int kb = k0 + 16 * hi;
        v16h b0 = *(const v16h*)(brow0 + kb);
        v16h b1 = *(const v16h*)(brow1 + kb);
        v16h b2 = *(const v16h*)(brow2 + kb);
        v16h b3 = *(const v16h*)(brow3 + kb);
        if (k0 + 32 < K) {            // pull next K tiles toward the WGP
            __builtin_prefetch(arow0 + k0 + 32, 0, 3);
            __builtin_prefetch(arow1 + k0 + 32, 0, 3);
            __builtin_prefetch(brow0 + kb + 32, 0, 3);
        }
        acc00 = __builtin_amdgcn_wmma_f32_16x16x32_f16(false, a0.v, false, b0,
                                                       (short)0, acc00, false, false);
        acc10 = __builtin_amdgcn_wmma_f32_16x16x32_f16(false, a1.v, false, b0,
                                                       (short)0, acc10, false, false);
        acc01 = __builtin_amdgcn_wmma_f32_16x16x32_f16(false, a0.v, false, b1,
                                                       (short)0, acc01, false, false);
        acc11 = __builtin_amdgcn_wmma_f32_16x16x32_f16(false, a1.v, false, b1,
                                                       (short)0, acc11, false, false);
        acc02 = __builtin_amdgcn_wmma_f32_16x16x32_f16(false, a0.v, false, b2,
                                                       (short)0, acc02, false, false);
        acc12 = __builtin_amdgcn_wmma_f32_16x16x32_f16(false, a1.v, false, b2,
                                                       (short)0, acc12, false, false);
        acc03 = __builtin_amdgcn_wmma_f32_16x16x32_f16(false, a0.v, false, b3,
                                                       (short)0, acc03, false, false);
        acc13 = __builtin_amdgcn_wmma_f32_16x16x32_f16(false, a1.v, false, b3,
                                                       (short)0, acc13, false, false);
    }

    const int col0 = n0 + r;
    const float bi0 = bias ? bias[col0]      : 0.0f;
    const float bi1 = bias ? bias[col0 + 16] : 0.0f;
    const float bi2 = bias ? bias[col0 + 32] : 0.0f;
    const float bi3 = bias ? bias[col0 + 48] : 0.0f;
#pragma unroll
    for (int rr = 0; rr < 8; ++rr) {
        float* crow0 = C + (long)(m0 + rr + 8 * hi) * ldc;
        crow0[col0]      = acc00[rr] + bi0;
        crow0[col0 + 16] = acc01[rr] + bi1;
        crow0[col0 + 32] = acc02[rr] + bi2;
        crow0[col0 + 48] = acc03[rr] + bi3;
        float* crow1 = C + (long)(m0 + 16 + rr + 8 * hi) * ldc;
        crow1[col0]      = acc10[rr] + bi0;
        crow1[col0 + 16] = acc11[rr] + bi1;
        crow1[col0 + 32] = acc12[rr] + bi2;
        crow1[col0 + 48] = acc13[rr] + bi3;
    }
}

// ---------------------------------------------------------------------------
// Softmax over k (Sk=512) with scale + rel-pos bias; writes f16 probabilities.
// grid: (Sq=512, Z) ; block: 256 (2 elements / thread)
// ---------------------------------------------------------------------------
__global__ __launch_bounds__(256) void softmax_rel_kernel(
    const float* __restrict__ scores, const float* __restrict__ rel,
    _Float16* __restrict__ probs, float scale)
{
    __shared__ float red[256];
    const int q = blockIdx.x;
    const long base = ((long)blockIdx.y * 512 + q) * 512;
    const float* srow = scores + base;
    const float* rrow = rel + (long)q * 512;
    const int t = threadIdx.x;

    float x0 = srow[t]       * scale + rrow[t];
    float x1 = srow[t + 256] * scale + rrow[t + 256];
    red[t] = fmaxf(x0, x1); __syncthreads();
    for (int s = 128; s > 0; s >>= 1) { if (t < s) red[t] = fmaxf(red[t], red[t + s]); __syncthreads(); }
    const float mx = red[0]; __syncthreads();

    float e0 = __expf(x0 - mx), e1 = __expf(x1 - mx);
    red[t] = e0 + e1; __syncthreads();
    for (int s = 128; s > 0; s >>= 1) { if (t < s) red[t] += red[t + s]; __syncthreads(); }
    const float inv = 1.0f / red[0];

    probs[base + t]       = (_Float16)(e0 * inv);
    probs[base + t + 256] = (_Float16)(e1 * inv);
}

// ---------------------------------------------------------------------------
// LayerNorm over F=768; optional f32 and f16 outputs. grid: (rows), block 256.
// ---------------------------------------------------------------------------
__global__ __launch_bounds__(256) void layernorm_kernel(
    const float* __restrict__ X, const float* __restrict__ w, const float* __restrict__ b,
    float* __restrict__ y32, _Float16* __restrict__ y16)
{
    __shared__ float red[256];
    const float* x = X + (long)blockIdx.x * 768;
    const int t = threadIdx.x;
    float a0 = x[t], a1 = x[t + 256], a2 = x[t + 512];
    red[t] = a0 + a1 + a2; __syncthreads();
    for (int s = 128; s > 0; s >>= 1) { if (t < s) red[t] += red[t + s]; __syncthreads(); }
    const float mean = red[0] * (1.0f / 768.0f); __syncthreads();
    float d0 = a0 - mean, d1 = a1 - mean, d2 = a2 - mean;
    red[t] = d0 * d0 + d1 * d1 + d2 * d2; __syncthreads();
    for (int s = 128; s > 0; s >>= 1) { if (t < s) red[t] += red[t + s]; __syncthreads(); }
    const float rstd = rsqrtf(red[0] * (1.0f / 768.0f) + 1e-5f);
    float r0 = d0 * rstd * w[t]       + b[t];
    float r1 = d1 * rstd * w[t + 256] + b[t + 256];
    float r2 = d2 * rstd * w[t + 512] + b[t + 512];
    if (y32) { float* o = y32 + (long)blockIdx.x * 768; o[t] = r0; o[t + 256] = r1; o[t + 512] = r2; }
    if (y16) { _Float16* o = y16 + (long)blockIdx.x * 768;
               o[t] = (_Float16)r0; o[t + 256] = (_Float16)r1; o[t + 512] = (_Float16)r2; }
}

// ------------------------------ elementwise --------------------------------
__global__ void cast_f32_to_f16_kernel(const float* __restrict__ x,
                                       _Float16* __restrict__ y, long n)
{
    long i = (long)blockIdx.x * blockDim.x + threadIdx.x;
    if (i < n) y[i] = (_Float16)x[i];
}

__global__ void transpose_cast_kernel(const float* __restrict__ W,
                                      _Float16* __restrict__ Wt, int Kd, int Nd)
{
    long i = (long)blockIdx.x * blockDim.x + threadIdx.x;
    if (i >= (long)Kd * Nd) return;
    int k = (int)(i / Nd), n = (int)(i % Nd);
    Wt[(long)n * Kd + k] = (_Float16)W[i];
}

// V f32 [8192 x 256] -> Vt f16 [B=16][H=4][DH=64][S=512]
__global__ void vtrans_kernel(const float* __restrict__ V, _Float16* __restrict__ Vt)
{
    long i = (long)blockIdx.x * blockDim.x + threadIdx.x;
    if (i >= (long)8192 * 256) return;
    int m = (int)(i >> 8), c = (int)(i & 255);
    int bb = m >> 9, s = m & 511;
    int h = c >> 6,  d = c & 63;
    Vt[((((long)bb * 4 + h) * 64 + d) << 9) + s] = (_Float16)V[i];
}

// cat[:, coloff:coloff+768] = f16( ln + gate * att )
__global__ void residual_gate_kernel(const float* __restrict__ ln,
                                     const float* __restrict__ att,
                                     const float* __restrict__ gate,
                                     _Float16* __restrict__ cat, int coloff)
{
    long i = (long)blockIdx.x * blockDim.x + threadIdx.x;
    if (i >= (long)8192 * 768) return;
    long m = i / 768; int c = (int)(i % 768);
    cat[m * 1536 + coloff + c] = (_Float16)(ln[i] + gate[0] * att[i]);
}

__global__ void silu_cast_kernel(const float* __restrict__ x,
                                 _Float16* __restrict__ y, long n)
{
    long i = (long)blockIdx.x * blockDim.x + threadIdx.x;
    if (i < n) { float v = x[i]; y[i] = (_Float16)(v / (1.0f + __expf(-v))); }
}

// ---------------------------------------------------------------------------
extern "C" void kernel_launch(void* const* d_in, const int* in_sizes, int n_in,
                              void* d_out, int out_size, void* d_ws, size_t ws_size,
                              hipStream_t stream)
{
    (void)in_sizes; (void)n_in; (void)out_size; (void)ws_size;
    const int  Mrows = 16 * 512;     // 8192
    const int  Hh = 4, CB = 4;       // heads, batch-chunk for attention
    const float SCALE = 0.125f;      // 1/sqrt(64)

    // ---- workspace carve-out (256B aligned) ----
    char* wsb = (char*)d_ws; size_t off = 0;
    auto alloc = [&](size_t bytes) -> void* {
        void* p = wsb + off;
        off = (off + bytes + 255) & ~((size_t)255);
        return p;
    };
    _Float16* w_rgb_t  = (_Float16*)alloc((size_t)768 * 768 * 2);
    _Float16* w_nir_t  = (_Float16*)alloc((size_t)768 * 768 * 2);
    _Float16* r2n_wq_t = (_Float16*)alloc((size_t)768 * 256 * 2);
    _Float16* r2n_wk_t = (_Float16*)alloc((size_t)768 * 256 * 2);
    _Float16* r2n_wv_t = (_Float16*)alloc((size_t)768 * 256 * 2);
    _Float16* r2n_wo_t = (_Float16*)alloc((size_t)768 * 256 * 2);
    _Float16* n2r_wq_t = (_Float16*)alloc((size_t)768 * 256 * 2);
    _Float16* n2r_wk_t = (_Float16*)alloc((size_t)768 * 256 * 2);
    _Float16* n2r_wv_t = (_Float16*)alloc((size_t)768 * 256 * 2);
    _Float16* n2r_wo_t = (_Float16*)alloc((size_t)768 * 256 * 2);
    _Float16* fw1_t    = (_Float16*)alloc((size_t)1536 * 768 * 2);
    _Float16* fw2_t    = (_Float16*)alloc((size_t)768 * 768 * 2);
    _Float16* xf16     = (_Float16*)alloc((size_t)Mrows * 768 * 2);
    float*    tmp      = (float*)   alloc((size_t)Mrows * 768 * 4);
    float*    rgb_ln32 = (float*)   alloc((size_t)Mrows * 768 * 4);
    _Float16* rgb_ln16 = (_Float16*)alloc((size_t)Mrows * 768 * 2);
    float*    nir_ln32 = (float*)   alloc((size_t)Mrows * 768 * 4);
    _Float16* nir_ln16 = (_Float16*)alloc((size_t)Mrows * 768 * 2);
    _Float16* qf       = (_Float16*)alloc((size_t)Mrows * 256 * 2);
    _Float16* kf       = (_Float16*)alloc((size_t)Mrows * 256 * 2);
    _Float16* vt       = (_Float16*)alloc((size_t)Mrows * 256 * 2);
    float*    scores   = (float*)   alloc((size_t)CB * Hh * 512 * 512 * 4);
    _Float16* probs    = (_Float16*)alloc((size_t)CB * Hh * 512 * 512 * 2);
    _Float16* cat      = (_Float16*)alloc((size_t)Mrows * 1536 * 2);

    // ---- launch helpers ----
    auto gemm = [&](const _Float16* A, const _Float16* Bt, float* C, const float* bias,
                    int lda, int ldb, int ldc, int M, int N, int K,
                    long sAb, long sAh, long sBb, long sBh, long sCb, long sCh,
                    int Hn, int Z) {
        dim3 grid((unsigned)((N + 127) / 128), (unsigned)((M + 127) / 128), (unsigned)Z);
        wmma_gemm_kernel<<<grid, dim3(256), 0, stream>>>(
            A, Bt, C, bias, lda, ldb, ldc, M, N, K,
            sAb, sAh, sBb, sBh, sCb, sCh, Hn);
    };
    auto tcast = [&](const void* W, _Float16* Wt, int Kd, int Nd) {
        long n = (long)Kd * Nd;
        transpose_cast_kernel<<<dim3((unsigned)((n + 255) / 256)), dim3(256), 0, stream>>>(
            (const float*)W, Wt, Kd, Nd);
    };
    auto cast16 = [&](const float* src, _Float16* dst, long n) {
        cast_f32_to_f16_kernel<<<dim3((unsigned)((n + 255) / 256)), dim3(256), 0, stream>>>(src, dst, n);
    };

    // ---- 0) weights -> transposed f16 ----
    tcast(d_in[2],  w_rgb_t, 768, 768);
    tcast(d_in[4],  w_nir_t, 768, 768);
    tcast(d_in[12], r2n_wq_t, 768, 256);
    tcast(d_in[13], r2n_wk_t, 768, 256);
    tcast(d_in[14], r2n_wv_t, 768, 256);
    tcast(d_in[15], r2n_wo_t, 256, 768);
    tcast(d_in[18], n2r_wq_t, 768, 256);
    tcast(d_in[19], n2r_wk_t, 768, 256);
    tcast(d_in[20], n2r_wv_t, 768, 256);
    tcast(d_in[21], n2r_wo_t, 256, 768);
    tcast(d_in[24], fw1_t, 1536, 768);
    tcast(d_in[26], fw2_t, 768, 768);

    // ---- 1) modality projections + LayerNorm ----
    cast16((const float*)d_in[0], xf16, (long)Mrows * 768);
    gemm(xf16, w_rgb_t, tmp, (const float*)d_in[3], 768, 768, 768, Mrows, 768, 768,
         0, 0, 0, 0, 0, 0, 1, 1);
    layernorm_kernel<<<dim3((unsigned)Mrows), dim3(256), 0, stream>>>(
        tmp, (const float*)d_in[6], (const float*)d_in[7], rgb_ln32, rgb_ln16);

    cast16((const float*)d_in[1], xf16, (long)Mrows * 768);
    gemm(xf16, w_nir_t, tmp, (const float*)d_in[5], 768, 768, 768, Mrows, 768, 768,
         0, 0, 0, 0, 0, 0, 1, 1);
    layernorm_kernel<<<dim3((unsigned)Mrows), dim3(256), 0, stream>>>(
        tmp, (const float*)d_in[8], (const float*)d_in[9], nir_ln32, nir_ln16);

    // ---- 2) bidirectional cross attention ----
    auto run_attn = [&](const _Float16* q_src, const _Float16* kv_src,
                        const _Float16* wq_t, const _Float16* wk_t, const _Float16* wv_t,
                        const _Float16* wo_t, const float* bo, const float* rel,
                        const float* ln_resid, const float* gate, int coloff) {
        // Q / K / V projections (f32 -> f16)
        gemm(q_src,  wq_t, tmp, nullptr, 768, 768, 256, Mrows, 256, 768, 0,0,0,0,0,0, 1, 1);
        cast16(tmp, qf, (long)Mrows * 256);
        gemm(kv_src, wk_t, tmp, nullptr, 768, 768, 256, Mrows, 256, 768, 0,0,0,0,0,0, 1, 1);
        cast16(tmp, kf, (long)Mrows * 256);
        gemm(kv_src, wv_t, tmp, nullptr, 768, 768, 256, Mrows, 256, 768, 0,0,0,0,0,0, 1, 1);
        vtrans_kernel<<<dim3((unsigned)(((long)Mrows * 256 + 255) / 256)), dim3(256), 0, stream>>>(tmp, vt);

        // attention in chunks of CB batches to bound scratch
        for (int cb = 0; cb < 16 / CB; ++cb) {
            const _Float16* qc = qf + (long)cb * CB * 512 * 256;
            const _Float16* kc = kf + (long)cb * CB * 512 * 256;
            // scores[z][q][k] = Q_bh * K_bh^T   (z = zb*H + zh)
            gemm(qc, kc, scores, nullptr, 256, 256, 512, 512, 512, 64,
                 (long)512 * 256, 64, (long)512 * 256, 64,
                 (long)Hh * 512 * 512, (long)512 * 512, Hh, CB * Hh);
            softmax_rel_kernel<<<dim3(512, (unsigned)(CB * Hh)), dim3(256), 0, stream>>>(
                scores, rel, probs, SCALE);
            // out[b*512+q, h*64+d] = probs * V
            gemm(probs, vt + (long)cb * CB * Hh * 64 * 512,
                 tmp + (long)cb * CB * 512 * 256, nullptr,
                 512, 512, 256, 512, 64, 512,
                 (long)Hh * 512 * 512, (long)512 * 512,
                 (long)Hh * 64 * 512,  (long)64 * 512,
                 (long)512 * 256, 64, Hh, CB * Hh);
        }
        // output projection + gated residual into concat buffer
        cast16(tmp, qf, (long)Mrows * 256);                    // reuse qf as attn f16
        gemm(qf, wo_t, tmp, bo, 256, 256, 768, Mrows, 768, 256, 0,0,0,0,0,0, 1, 1);
        residual_gate_kernel<<<dim3((unsigned)(((long)Mrows * 768 + 255) / 256)), dim3(256), 0, stream>>>(
            ln_resid, tmp, gate, cat, coloff);
    };

    run_attn(rgb_ln16, nir_ln16, r2n_wq_t, r2n_wk_t, r2n_wv_t, r2n_wo_t,
             (const float*)d_in[16], (const float*)d_in[17],
             rgb_ln32, (const float*)d_in[28], 0);
    run_attn(nir_ln16, rgb_ln16, n2r_wq_t, n2r_wk_t, n2r_wv_t, n2r_wo_t,
             (const float*)d_in[22], (const float*)d_in[23],
             nir_ln32, (const float*)d_in[29], 768);

    // ---- 3) fusion MLP + final LayerNorm ----
    gemm(cat, fw1_t, tmp, (const float*)d_in[25], 1536, 1536, 768, Mrows, 768, 1536,
         0, 0, 0, 0, 0, 0, 1, 1);
    silu_cast_kernel<<<dim3((unsigned)(((long)Mrows * 768 + 255) / 256)), dim3(256), 0, stream>>>(
        tmp, xf16, (long)Mrows * 768);
    gemm(xf16, fw2_t, tmp, (const float*)d_in[27], 768, 768, 768, Mrows, 768, 768,
         0, 0, 0, 0, 0, 0, 1, 1);
    layernorm_kernel<<<dim3((unsigned)Mrows), dim3(256), 0, stream>>>(
        tmp, (const float*)d_in[10], (const float*)d_in[11], (float*)d_out, nullptr);
}
